// LSTMBaseline_4638564680411
// MI455X (gfx1250) — matrix operand order repaired and checked by reading, not compiled, not run on hardware
//
#include <hip/hip_runtime.h>

typedef __attribute__((ext_vector_type(16))) _Float16 v16h;
typedef __attribute__((ext_vector_type(8)))  _Float16 v8h;
typedef __attribute__((ext_vector_type(8)))  float    v8f;

#define LSTM_B 512
#define LSTM_T 1024
#define LSTM_D 4
#define LSTM_H 64

#define BT     16   // batch rows per workgroup
#define NWAVES 4    // one wave per 16-wide hidden slice (4*16 = 64 = H)

// gfx1250 has a native V_TANH_F32 trans op; use it when the builtin exists.
#if __has_builtin(__builtin_amdgcn_tanhf)
__device__ __forceinline__ float tanh_f(float x) { return __builtin_amdgcn_tanhf(x); }
#else
__device__ __forceinline__ float tanh_f(float x) { return 1.0f - 2.0f / (1.0f + __expf(2.0f * x)); }
#endif
// sigmoid(x) = 0.5*tanh(x/2) + 0.5  (one trans op + one FMA)
__device__ __forceinline__ float sigm(float x) { return fmaf(tanh_f(0.5f * x), 0.5f, 0.5f); }

// Load one A-operand K-chunk (16x32 f16) for this lane from an LDS [16][64] f16 buffer.
// A layout (wave32): M = lane&15; element e -> K = e + 16*(lane>>4) + kc.
__device__ __forceinline__ v16h load_A_chunk(const _Float16* buf, int m, int khi, int kc) {
  const v8h lo = *reinterpret_cast<const v8h*>(buf + m * LSTM_H + kc + khi);
  const v8h hi = *reinterpret_cast<const v8h*>(buf + m * LSTM_H + kc + khi + 8);
  return __builtin_shufflevector(lo, hi, 0,1,2,3,4,5,6,7,8,9,10,11,12,13,14,15);
}

__global__ __launch_bounds__(BT * 8, 1)
void lstm2_wmma_kernel(const float* __restrict__ x,
                       const float* __restrict__ W_ih0, const float* __restrict__ W_hh0,
                       const float* __restrict__ b0,
                       const float* __restrict__ W_ih1, const float* __restrict__ W_hh1,
                       const float* __restrict__ b1,
                       const float* __restrict__ W_fc1, const float* __restrict__ b_fc1,
                       const float* __restrict__ W_fc2, const float* __restrict__ b_fc2,
                       float* __restrict__ out)
{
  __shared__ _Float16 hbuf1[BT * LSTM_H];   // layer-0 hidden state, f16, [M][K]
  __shared__ _Float16 hbuf2[BT * LSTM_H];   // layer-1 hidden state, f16, [M][K]
  __shared__ float    h2f  [BT * LSTM_H];   // final h2 in f32 for the classifier

  const int tid    = threadIdx.x;
  const int wave   = tid >> 5;          // hidden slice: columns [16*wave, 16*wave+16)
  const int lane   = tid & 31;
  const int lanelo = lane & 15;
  const int lanehi = lane >> 4;         // 0 or 1
  const int khiA   = lanehi * 16;       // A-operand K bias
  const int kbB    = lanehi * 16;       // B-operand K bias
  const int mrow   = lanehi * 8;        // C/D accumulator row base
  const int bb     = blockIdx.x * BT;   // batch base for this workgroup
  const float lmask = (lanehi == 0) ? 1.0f : 0.0f;

  // ---------------- weights -> registers (B-operand layout, f16) ----------------
  // B layout (32x16 chunk): N = lane&15, element e -> K = kbB + e (within the chunk).
  v16h Bhh0[4][2], Bih1[4][2], Bhh1[4][2], Bx[4];
  float b0v[4], b1v[4];
#pragma unroll
  for (int g = 0; g < 4; ++g) {
    const int ng = g * LSTM_H + wave * 16 + lanelo;   // global gate column owned by this lane
    b0v[g] = b0[ng];
    b1v[g] = b1[ng];
#pragma unroll
    for (int c = 0; c < 2; ++c) {
      const float* p0 = W_hh0 + ng * LSTM_H + c * 32 + kbB;
      const float* p1 = W_ih1 + ng * LSTM_H + c * 32 + kbB;
      const float* p2 = W_hh1 + ng * LSTM_H + c * 32 + kbB;
      v16h t0, t1, t2;
#pragma unroll
      for (int e = 0; e < 16; ++e) {
        t0[e] = (_Float16)p0[e];
        t1[e] = (_Float16)p1[e];
        t2[e] = (_Float16)p2[e];
      }
      Bhh0[g][c] = t0; Bih1[g][c] = t1; Bhh1[g][c] = t2;
    }
    // x-weights: only K=0..3 live (lanes 0-15), rest zero-padded
    v16h bx;
#pragma unroll
    for (int e = 0; e < 16; ++e) bx[e] = (_Float16)0.0f;
#pragma unroll
    for (int e = 0; e < LSTM_D; ++e) bx[e] = (_Float16)(W_ih0[ng * LSTM_D + e] * lmask);
    Bx[g] = bx;
  }

  // ---------------- recurrent state ----------------
  v16h Ah1[2], Ah2[2];
#pragma unroll
  for (int c = 0; c < 2; ++c) {
    v16h z;
#pragma unroll
    for (int e = 0; e < 16; ++e) z[e] = (_Float16)0.0f;
    Ah1[c] = z; Ah2[c] = z;
  }
  v8f c1, c2, h2t;
#pragma unroll
  for (int r = 0; r < 8; ++r) { c1[r] = 0.0f; c2[r] = 0.0f; h2t[r] = 0.0f; }

  const int ncol = wave * 16 + lanelo;
  const float* xrow = x + (size_t)(bb + lanelo) * LSTM_T * LSTM_D;   // this lane's batch row

  // Software pipeline: preload x_0; each step issues the load for t+1 up front
  // so global latency overlaps the whole step's compute.
  float4 xv = *reinterpret_cast<const float4*>(xrow);

  // ---------------- time loop (strictly sequential recurrence) ----------------
  for (int t = 0; t < LSTM_T; ++t) {
    const int tn = (t + 1 < LSTM_T) ? (t + 1) : t;
    const float4 xv_next = *reinterpret_cast<const float4*>(xrow + (size_t)tn * LSTM_D);
    {
      const int tp = (t + 8 < LSTM_T) ? (t + 8) : (LSTM_T - 1);
      __builtin_prefetch(xrow + (size_t)tp * LSTM_D, 0, 0);
    }

    // x_t -> zero-padded A tile (K=0..3 live on lanes 0-15)
    v16h Ax;
#pragma unroll
    for (int e = 0; e < 16; ++e) Ax[e] = (_Float16)0.0f;
    Ax[0] = (_Float16)(xv.x * lmask);
    Ax[1] = (_Float16)(xv.y * lmask);
    Ax[2] = (_Float16)(xv.z * lmask);
    Ax[3] = (_Float16)(xv.w * lmask);

    // ---- layer 0: pre = b0 + x@W_ih0^T + h1@W_hh0^T ----
    v8f acc[4];
#pragma unroll
    for (int g = 0; g < 4; ++g) {
      v8f a;
#pragma unroll
      for (int r = 0; r < 8; ++r) a[r] = b0v[g];
      a = __builtin_amdgcn_wmma_f32_16x16x32_f16(false, Ax,     false, Bx[g],      (short)0, a, false, false);
      a = __builtin_amdgcn_wmma_f32_16x16x32_f16(false, Ah1[0], false, Bhh0[g][0], (short)0, a, false, false);
      a = __builtin_amdgcn_wmma_f32_16x16x32_f16(false, Ah1[1], false, Bhh0[g][1], (short)0, a, false, false);
      acc[g] = a;
    }
    v8f h1t;
#pragma unroll
    for (int r = 0; r < 8; ++r) {
      const float iv = sigm(acc[0][r]);
      const float fv = sigm(acc[1][r]);
      const float gv = tanh_f(acc[2][r]);
      const float ov = sigm(acc[3][r]);
      const float c  = fv * c1[r] + iv * gv;
      c1[r]  = c;
      h1t[r] = ov * tanh_f(c);
    }

    // share h1 across waves (prev readers of hbuf1 are fenced by last iter's barrier)
#pragma unroll
    for (int r = 0; r < 8; ++r)
      hbuf1[(mrow + r) * LSTM_H + ncol] = (_Float16)h1t[r];
    __syncthreads();
    Ah1[0] = load_A_chunk(hbuf1, lanelo, khiA, 0);
    Ah1[1] = load_A_chunk(hbuf1, lanelo, khiA, 32);

    // ---- layer 1: pre = b1 + h1@W_ih1^T + h2@W_hh1^T ----
#pragma unroll
    for (int g = 0; g < 4; ++g) {
      v8f a;
#pragma unroll
      for (int r = 0; r < 8; ++r) a[r] = b1v[g];
      a = __builtin_amdgcn_wmma_f32_16x16x32_f16(false, Ah1[0], false, Bih1[g][0], (short)0, a, false, false);
      a = __builtin_amdgcn_wmma_f32_16x16x32_f16(false, Ah1[1], false, Bih1[g][1], (short)0, a, false, false);
      a = __builtin_amdgcn_wmma_f32_16x16x32_f16(false, Ah2[0], false, Bhh1[g][0], (short)0, a, false, false);
      a = __builtin_amdgcn_wmma_f32_16x16x32_f16(false, Ah2[1], false, Bhh1[g][1], (short)0, a, false, false);
      acc[g] = a;
    }
#pragma unroll
    for (int r = 0; r < 8; ++r) {
      const float iv = sigm(acc[0][r]);
      const float fv = sigm(acc[1][r]);
      const float gv = tanh_f(acc[2][r]);
      const float ov = sigm(acc[3][r]);
      const float c  = fv * c2[r] + iv * gv;
      c2[r]  = c;
      h2t[r] = ov * tanh_f(c);
    }

#pragma unroll
    for (int r = 0; r < 8; ++r)
      hbuf2[(mrow + r) * LSTM_H + ncol] = (_Float16)h2t[r];
    __syncthreads();
    Ah2[0] = load_A_chunk(hbuf2, lanelo, khiA, 0);
    Ah2[1] = load_A_chunk(hbuf2, lanelo, khiA, 32);

    xv = xv_next;   // rotate the software pipeline
  }

  // ---------------- classifier head on last h2 ----------------
#pragma unroll
  for (int r = 0; r < 8; ++r)
    h2f[(mrow + r) * LSTM_H + ncol] = h2t[r];
  __syncthreads();

  if (tid < BT) {
    const int m = tid;
    float o = b_fc2[0];
#pragma unroll 4
    for (int j = 0; j < 32; ++j) {
      float s = b_fc1[j];
#pragma unroll 8
      for (int k = 0; k < LSTM_H; ++k)
        s = fmaf(h2f[m * LSTM_H + k], W_fc1[j * LSTM_H + k], s);
      s = fmaxf(s, 0.0f);
      o = fmaf(s, W_fc2[j], o);
    }
    out[bb + m] = sigm(o);
  }
}

extern "C" void kernel_launch(void* const* d_in, const int* in_sizes, int n_in,
                              void* d_out, int out_size, void* d_ws, size_t ws_size,
                              hipStream_t stream) {
  const float* x     = (const float*)d_in[0];
  const float* W_ih0 = (const float*)d_in[1];
  const float* W_hh0 = (const float*)d_in[2];
  const float* b0    = (const float*)d_in[3];
  const float* W_ih1 = (const float*)d_in[4];
  const float* W_hh1 = (const float*)d_in[5];
  const float* b1    = (const float*)d_in[6];
  const float* W_fc1 = (const float*)d_in[7];
  const float* b_fc1 = (const float*)d_in[8];
  const float* W_fc2 = (const float*)d_in[9];
  const float* b_fc2 = (const float*)d_in[10];
  float* out = (float*)d_out;

  dim3 grid(LSTM_B / BT);     // 32 workgroups (one 16-row batch tile each)
  dim3 block(BT * 8);         // 128 threads = 4 wave32s (one per hidden slice)
  hipLaunchKernelGGL(lstm2_wmma_kernel, grid, block, 0, stream,
                     x, W_ih0, W_hh0, b0, W_ih1, W_hh1, b1,
                     W_fc1, b_fc1, W_fc2, b_fc2, out);
}
